// Attention_38233798869191
// MI455X (gfx1250) — compile-verified, matrix operand order checked by reading, not run on hardware
//
#include <hip/hip_runtime.h>
#include <hip/hip_bf16.h>

// ---------------------------------------------------------------------------
// Multi-head attention forward for MI455X (gfx1250), wave32 + WMMA f16.
// B=2, S=2048, D=1024, H=16, DH=64, SCALE = 1/8.
// ---------------------------------------------------------------------------

#define B_  2
#define S_  2048
#define D_  1024
#define H_  16
#define DH_ 64
#define SCALE_ 0.125f

typedef __attribute__((ext_vector_type(16))) _Float16 v16h;
typedef __attribute__((ext_vector_type(8)))  _Float16 v8h;
typedef __attribute__((ext_vector_type(8)))  float    v8f;

// D = A(16x32 f16) * B(32x16 f16) + C(16x16 f32)
__device__ __forceinline__ v8f wmma_f16(v16h a, v16h b, v8f c) {
  return __builtin_amdgcn_wmma_f32_16x16x32_f16(
      false, a, false, b, (short)0, c, false, false);
}

// Load one A/B fragment (16 halves) for v_wmma_f32_16x16x32_f16.
// A layout (ISA 7.12.2): lanes 0-15 -> row, K {0..7,16..23}; lanes 16-31 ->
// same rows, K {8..15,24..31}.  B is mirrored with lane = N column, reading
// the "row" of the K-major source.  `row` must already include (lane & 15).
__device__ __forceinline__ v16h load_frag(const _Float16* base, int stride,
                                          int row, int kbase, int half) {
  const _Float16* p = base + (size_t)row * stride + kbase + half * 8;
  v8h lo = *(const v8h*)p;          // K +0..7   (or +8..15)
  v8h hi = *(const v8h*)(p + 16);   // K +16..23 (or +24..31)
  v16h r;
#pragma unroll
  for (int i = 0; i < 8; ++i) { r[i] = lo[i]; r[i + 8] = hi[i]; }
  return r;
}

// ---------------------------------------------------------------------------
// f32 -> f16 conversion
// ---------------------------------------------------------------------------
__global__ void cvt_f32_to_f16(const float* __restrict__ in,
                               _Float16* __restrict__ out, int n) {
  int i = blockIdx.x * blockDim.x + threadIdx.x;
  if (i < n) out[i] = (_Float16)in[i];
}

// ---------------------------------------------------------------------------
// Generic WMMA GEMM:  y[m,n] = sum_k A[m,k] * W[n,k] + bias[n]
// A: [M x 1024] f16 row-major, W: [1024 x 1024] f16 row-major (torch Linear).
// One wave computes a 16x64 output tile; the K-loop is software-pipelined so
// the next K-step's 10 b128 loads are in flight while the current 4 WMMAs
// issue (keeps the matrix pipe from stalling on s_wait_loadcnt 0).
// mode 0: q -> scale by SCALE_, store f16 [B,H,S,DH]
// mode 1: k ->                 store f16 [B,H,S,DH]
// mode 2: v ->                 store f16 [B,H,DH,S]  (transposed for P@V)
// mode 3: final out -> f32 row-major [M x 1024]
// ---------------------------------------------------------------------------
__global__ __launch_bounds__(256)
void gemm_wmma_16x64(const _Float16* __restrict__ A,
                     const _Float16* __restrict__ W,
                     const float* __restrict__ bias,
                     void* __restrict__ outp, int M, int mode) {
  const int wave = (blockIdx.x * blockDim.x + threadIdx.x) >> 5;
  const int lane = threadIdx.x & 31;
  const int half = lane >> 4;
  const int ln   = lane & 15;

  const int NT64 = D_ / 64;              // 16 column-supertiles
  const int mt = wave / NT64;
  const int nt = wave % NT64;
  if (mt * 16 >= M) return;
  const int row0 = mt * 16;
  const int col0 = nt * 64;

  v8f c0 = {}, c1 = {}, c2 = {}, c3 = {};

  // ---- software-pipelined K loop: prologue loads K-step 0 ----
  v16h a  = load_frag(A, D_, row0 + ln,       0, half);
  v16h b0 = load_frag(W, D_, col0 +  0 + ln,  0, half);
  v16h b1 = load_frag(W, D_, col0 + 16 + ln,  0, half);
  v16h b2 = load_frag(W, D_, col0 + 32 + ln,  0, half);
  v16h b3 = load_frag(W, D_, col0 + 48 + ln,  0, half);

  for (int kb = 0; kb < D_ - 32; kb += 32) {
    // issue next K-step's loads before consuming the current fragments
    const int kn = kb + 32;
    v16h an  = load_frag(A, D_, row0 + ln,      kn, half);
    v16h bn0 = load_frag(W, D_, col0 +  0 + ln, kn, half);
    v16h bn1 = load_frag(W, D_, col0 + 16 + ln, kn, half);
    v16h bn2 = load_frag(W, D_, col0 + 32 + ln, kn, half);
    v16h bn3 = load_frag(W, D_, col0 + 48 + ln, kn, half);

    c0 = wmma_f16(a, b0, c0);
    c1 = wmma_f16(a, b1, c1);
    c2 = wmma_f16(a, b2, c2);
    c3 = wmma_f16(a, b3, c3);

    a = an; b0 = bn0; b1 = bn1; b2 = bn2; b3 = bn3;
  }
  // epilogue K-step
  c0 = wmma_f16(a, b0, c0);
  c1 = wmma_f16(a, b1, c1);
  c2 = wmma_f16(a, b2, c2);
  c3 = wmma_f16(a, b3, c3);

  // Epilogue.  C layout: VGPR i -> row i + 8*half, column = lane & 15.
#pragma unroll
  for (int i = 0; i < 8; ++i) {
    const int m = row0 + i + half * 8;       // global row in [0, B*S)
    const int bb = m >> 11;                  // / S_
    const int ss = m & (S_ - 1);
#pragma unroll
    for (int t = 0; t < 4; ++t) {
      const int n = col0 + t * 16 + ln;      // output feature
      float v = (t == 0 ? c0[i] : t == 1 ? c1[i] : t == 2 ? c2[i] : c3[i]);
      v += bias[n];
      if (mode == 3) {
        ((float*)outp)[(size_t)m * D_ + n] = v;
      } else {
        const int h = n >> 6;                // / DH_
        const int d = n & (DH_ - 1);
        const int bh = bb * H_ + h;
        _Float16* o16 = (_Float16*)outp;
        if (mode == 0) {
          o16[(((size_t)bh * S_ + ss) * DH_) + d] = (_Float16)(v * SCALE_);
        } else if (mode == 1) {
          o16[(((size_t)bh * S_ + ss) * DH_) + d] = (_Float16)v;
        } else {
          o16[(((size_t)bh * DH_ + d) * S_) + ss] = (_Float16)v;  // transposed
        }
      }
    }
  }
}

// ---------------------------------------------------------------------------
// Flash attention: one wave per (b, h, 16-query block).
// Q pre-scaled.  K: [B,H,S,64] f16.  Vt: [B,H,64,S] f16.
// Output scattered to att[B,S,D] f16 with the reference's faithful
// reshape(B,D,S).swapaxes(-1,-2) scramble.
// ---------------------------------------------------------------------------
__global__ __launch_bounds__(32)
void attn_flash(const _Float16* __restrict__ Q,
                const _Float16* __restrict__ K,
                const _Float16* __restrict__ Vt,
                _Float16* __restrict__ att) {
  __shared__ __align__(16) _Float16 pbuf[16 * 32];   // P tile transpose buffer

  const int lane = threadIdx.x & 31;
  const int half = lane >> 4;
  const int ln   = lane & 15;

  const int blk = blockIdx.x;            // B*H*(S/16) = 4096
  const int qb  = blk & (S_ / 16 - 1);   // 128 query blocks
  const int bh  = blk >> 7;              // b*H + h
  const int h   = bh & (H_ - 1);
  const int b   = bh >> 4;

  const _Float16* Qh = Q  + (size_t)bh * S_ * DH_;
  const _Float16* Kh = K  + (size_t)bh * S_ * DH_;
  const _Float16* Vh = Vt + (size_t)bh * DH_ * S_;

  // Q fragments for this 16-row block (K-dim = 64 -> two frags), fixed.
  const v16h q0 = load_frag(Qh, DH_, qb * 16 + ln, 0,  half);
  const v16h q1 = load_frag(Qh, DH_, qb * 16 + ln, 32, half);

  float m_[8], l_[8];
  v8f o0 = {}, o1 = {}, o2 = {}, o3 = {};
#pragma unroll
  for (int i = 0; i < 8; ++i) { m_[i] = -__builtin_inff(); l_[i] = 0.0f; }

  for (int jb = 0; jb < S_; jb += 32) {
    // ---- scores: 16x32 tile, K-dim 64 (two WMMA steps per 16-col tile) ----
    v8f s0 = {}, s1 = {};
    {
      v16h ka0 = load_frag(Kh, DH_, jb + ln,      0,  half);
      v16h ka1 = load_frag(Kh, DH_, jb + 16 + ln, 0,  half);
      v16h kb0 = load_frag(Kh, DH_, jb + ln,      32, half);
      v16h kb1 = load_frag(Kh, DH_, jb + 16 + ln, 32, half);
      s0 = wmma_f16(q0, ka0, s0);
      s1 = wmma_f16(q0, ka1, s1);
      s0 = wmma_f16(q1, kb0, s0);
      s1 = wmma_f16(q1, kb1, s1);
    }

    // ---- online softmax update (rows live at (vgpr i, 16-lane half)) ----
    float alpha[8];
#pragma unroll
    for (int i = 0; i < 8; ++i) {
      float mx = fmaxf(s0[i], s1[i]);
      mx = fmaxf(mx, __shfl_xor(mx, 1, 32));
      mx = fmaxf(mx, __shfl_xor(mx, 2, 32));
      mx = fmaxf(mx, __shfl_xor(mx, 4, 32));
      mx = fmaxf(mx, __shfl_xor(mx, 8, 32));
      const float mnew = fmaxf(m_[i], mx);
      const float a = __expf(m_[i] - mnew);
      alpha[i] = a;
      const float p0 = __expf(s0[i] - mnew);
      const float p1 = __expf(s1[i] - mnew);
      s0[i] = p0; s1[i] = p1;
      float sum = p0 + p1;
      sum += __shfl_xor(sum, 1, 32);
      sum += __shfl_xor(sum, 2, 32);
      sum += __shfl_xor(sum, 4, 32);
      sum += __shfl_xor(sum, 8, 32);
      l_[i] = l_[i] * a + sum;
      m_[i] = mnew;
    }

    // Rescale running output accumulators.
#pragma unroll
    for (int i = 0; i < 8; ++i) {
      o0[i] *= alpha[i]; o1[i] *= alpha[i];
      o2[i] *= alpha[i]; o3[i] *= alpha[i];
    }

    // ---- transpose P (C layout) -> A-fragment layout via LDS ----
#pragma unroll
    for (int i = 0; i < 8; ++i) {
      const int row = i + half * 8;
      pbuf[row * 32 + ln]      = (_Float16)s0[i];
      pbuf[row * 32 + ln + 16] = (_Float16)s1[i];
    }
    __syncthreads();                     // single wave: DS ops are in-order
    const v16h pa = load_frag(pbuf, 32, ln, 0, half);

    // ---- O += P(16x32) @ V(32x64) ----
    {
      v16h vb0 = load_frag(Vh, S_,  0 + ln, jb, half);
      v16h vb1 = load_frag(Vh, S_, 16 + ln, jb, half);
      v16h vb2 = load_frag(Vh, S_, 32 + ln, jb, half);
      v16h vb3 = load_frag(Vh, S_, 48 + ln, jb, half);
      o0 = wmma_f16(pa, vb0, o0);
      o1 = wmma_f16(pa, vb1, o1);
      o2 = wmma_f16(pa, vb2, o2);
      o3 = wmma_f16(pa, vb3, o3);
    }
    __syncthreads();
  }

  // ---- epilogue: normalize and scatter with the reference's scramble ----
#pragma unroll
  for (int i = 0; i < 8; ++i) {
    const float inv = 1.0f / l_[i];
    const int row = i + half * 8;
    const int s = qb * 16 + row;
    const int fbase = (h * S_ + s) * DH_;
#pragma unroll
    for (int t = 0; t < 4; ++t) {
      const int d = t * 16 + ln;
      const float v = (t == 0 ? o0[i] : t == 1 ? o1[i] : t == 2 ? o2[i] : o3[i]) * inv;
      const int f  = fbase + d;          // flat (h,s,d) index
      const int io = f >> 11;            // / S_
      const int jo = f & (S_ - 1);
      att[((size_t)b * S_ + jo) * D_ + io] = (_Float16)v;
    }
  }
}

// ---------------------------------------------------------------------------
// Host-side launcher
// ---------------------------------------------------------------------------
extern "C" void kernel_launch(void* const* d_in, const int* in_sizes, int n_in,
                              void* d_out, int out_size, void* d_ws, size_t ws_size,
                              hipStream_t stream) {
  const float* x  = (const float*)d_in[0];
  const float* Wq = (const float*)d_in[1];
  const float* bq = (const float*)d_in[2];
  const float* Wk = (const float*)d_in[3];
  const float* bk = (const float*)d_in[4];
  const float* Wv = (const float*)d_in[5];
  const float* bv = (const float*)d_in[6];
  const float* Wo = (const float*)d_in[7];
  const float* bo = (const float*)d_in[8];
  float* out = (float*)d_out;

  const int M  = B_ * S_;                // 4096
  const int NX = M * D_;                 // x elements: 4194304
  const int NW = D_ * D_;                // weight elements: 1048576

  // Workspace layout (f16 element offsets, all 16B-aligned):
  _Float16* ws  = (_Float16*)d_ws;
  _Float16* xh  = ws;                    // [M, D]
  _Float16* Wqh = xh  + (size_t)NX;      // [D, D]
  _Float16* Wkh = Wqh + (size_t)NW;
  _Float16* Wvh = Wkh + (size_t)NW;
  _Float16* Woh = Wvh + (size_t)NW;
  _Float16* Qh  = Woh + (size_t)NW;      // [B,H,S,DH], pre-scaled
  _Float16* Kh  = Qh  + (size_t)NX;      // [B,H,S,DH]
  _Float16* Vth = Kh  + (size_t)NX;      // [B,H,DH,S]
  _Float16* att = Vth + (size_t)NX;      // [B,S,D] scrambled attention out

  // 1) f32 -> f16 conversions
  cvt_f32_to_f16<<<(NX + 255) / 256, 256, 0, stream>>>(x,  xh,  NX);
  cvt_f32_to_f16<<<(NW + 255) / 256, 256, 0, stream>>>(Wq, Wqh, NW);
  cvt_f32_to_f16<<<(NW + 255) / 256, 256, 0, stream>>>(Wk, Wkh, NW);
  cvt_f32_to_f16<<<(NW + 255) / 256, 256, 0, stream>>>(Wv, Wvh, NW);
  cvt_f32_to_f16<<<(NW + 255) / 256, 256, 0, stream>>>(Wo, Woh, NW);

  // 2) QKV projections: (M/16)*(D/64) = 4096 waves, 8 waves/block
  const int gemm_blocks = (M / 16) * (D_ / 64) / 8;   // 512
  gemm_wmma_16x64<<<gemm_blocks, 256, 0, stream>>>(xh, Wqh, bq, Qh,  M, 0);
  gemm_wmma_16x64<<<gemm_blocks, 256, 0, stream>>>(xh, Wkh, bk, Kh,  M, 1);
  gemm_wmma_16x64<<<gemm_blocks, 256, 0, stream>>>(xh, Wvh, bv, Vth, M, 2);

  // 3) Flash attention: one wave per (b, h, 16-query block)
  attn_flash<<<B_ * H_ * (S_ / 16), 32, 0, stream>>>(Qh, Kh, Vth, att);

  // 4) Output projection -> f32 d_out
  gemm_wmma_16x64<<<gemm_blocks, 256, 0, stream>>>(att, Woh, bo, out, M, 3);
}
